// SpatialAttention_45243185496265
// MI455X (gfx1250) — matrix operand order repaired and checked by reading, not compiled, not run on hardware
//
#include <hip/hip_runtime.h>

// ---------------------------------------------------------------------------
// Types for CDNA5 WMMA
// ---------------------------------------------------------------------------
typedef __attribute__((ext_vector_type(16))) __bf16 v16bf;
typedef __attribute__((ext_vector_type(8)))  float  v8f;
typedef unsigned int       u32;
typedef unsigned short     u16;

union FragU {
    u32   u[8];
    v16bf v;
};

// float -> bf16 (round to nearest even, no NaN special-casing)
static __device__ __forceinline__ u16 f2bf(float x) {
    u32 u = __builtin_bit_cast(u32, x);
    u32 r = u + 0x7FFFu + ((u >> 16) & 1u);
    return (u16)(r >> 16);
}

// xor-butterfly within 16-lane halves. Prefer v_permlane16_b32 (VALU pipe,
// no LDS traffic) over ds_bpermute-based __shfl_xor.
static __device__ __forceinline__ float pl16(float v, u32 lo, u32 hi) {
#if __has_builtin(__builtin_amdgcn_permlane16)
    u32 s = __builtin_bit_cast(u32, v);
    u32 d = __builtin_amdgcn_permlane16(s, s, lo, hi, false, false);
    return __builtin_bit_cast(float, d);
#else
    (void)lo; (void)hi;
    return v;  // unused in fallback path
#endif
}

static __device__ __forceinline__ float rmax16(float v) {
#if __has_builtin(__builtin_amdgcn_permlane16)
    // selectors: nibble i of {hi,lo} = (lane i) ^ m, per 16-lane row
    v = fmaxf(v, pl16(v, 0x67452301u, 0xEFCDAB89u));  // xor 1
    v = fmaxf(v, pl16(v, 0x54761032u, 0xDCFE98BAu));  // xor 2
    v = fmaxf(v, pl16(v, 0x32107654u, 0xBA98FEDCu));  // xor 4
    v = fmaxf(v, pl16(v, 0xFEDCBA98u, 0x76543210u));  // xor 8
    return v;
#else
    #pragma unroll
    for (int m = 1; m < 16; m <<= 1) v = fmaxf(v, __shfl_xor(v, m, 32));
    return v;
#endif
}

// Problem constants
#define CC   64      // channels (head dim)
#define LL   1024    // sequence length
#define NBT  64      // B*T_OUT batches

// ---------------------------------------------------------------------------
// Kernel 1: QKV projection + pack into WMMA fragment layouts (bf16).
//
// Fragment buffers (dword units, each dword = packed bf16 pair):
//   qf: A-fragments of Q.  [bt][it(64)][cc(2)][lane(32)][v(8)]
//   kf: B-fragments of K^T.[bt][jt16(64)][cc(2)][lane(32)][v(8)]
//   vf: B-fragments of V^T.[bt][jt32(32)][h(4)][lane(32)][v(8)]
//
// Layouts per CDNA5 ISA 7.12.2 (wave32):
//   A 16x32 bf16: lane -> M = lane&15, half = lane>>4;
//                 v<4: K = 2v + 8*half ; v>=4: K = 16 + 2(v-4) + 8*half
//   B 32x16 bf16: lane -> N = lane&15, half = lane>>4; v<4: K = 2v + 8*half ; v>=4: K = 16 + 2(v-4) + 8*half
// ---------------------------------------------------------------------------
__global__ __launch_bounds__(256)
void qkv_pack_kernel(const float* __restrict__ x,
                     const float* __restrict__ Wq, const float* __restrict__ bq,
                     const float* __restrict__ Wk, const float* __restrict__ bk,
                     const float* __restrict__ Wv, const float* __restrict__ bv,
                     u32* __restrict__ qf, u32* __restrict__ kf, u32* __restrict__ vf)
{
    extern __shared__ char smem[];
    float* wqs = (float*)smem;                 // 64*64
    float* wks = wqs + 4096;
    float* wvs = wks + 4096;
    float* bqs = wvs + 4096;                   // 64 each
    float* bks = bqs + 64;
    float* bvs = bks + 64;
    float* xs  = bvs + 64;                     // [c(64)][j(64)]
    u16*   qls = (u16*)(xs + 4096);            // [lloc(64)][o(64)] bf16
    u16*   kls = qls + 4096;
    u16*   vls = kls + 4096;

    const int t     = threadIdx.x;
    const int bt    = blockIdx.x >> 4;
    const int chunk = blockIdx.x & 15;
    const int l0    = chunk * 64;

    for (int idx = t; idx < 4096; idx += 256) {
        wqs[idx] = Wq[idx]; wks[idx] = Wk[idx]; wvs[idx] = Wv[idx];
    }
    if (t < 64) { bqs[t] = bq[t]; bks[t] = bk[t]; bvs[t] = bv[t]; }
    for (int idx = t; idx < 4096; idx += 256) {
        int c = idx >> 6, j = idx & 63;
        xs[idx] = x[((bt * CC + c) << 10) + l0 + j];
    }
    __syncthreads();

    // each thread: one output channel o, 16 sequence positions
    const int o  = t >> 2;
    const int jb = (t & 3) << 4;
    float aq[16], ak[16], av[16];
    #pragma unroll
    for (int i = 0; i < 16; ++i) { aq[i] = bqs[o]; ak[i] = bks[o]; av[i] = bvs[o]; }
    for (int c = 0; c < CC; ++c) {
        float wq = wqs[o * 64 + c], wk = wks[o * 64 + c], wv = wvs[o * 64 + c];
        const float* xr = &xs[c * 64 + jb];
        #pragma unroll
        for (int i = 0; i < 16; ++i) {
            float xv = xr[i];
            aq[i] = fmaf(wq, xv, aq[i]);
            ak[i] = fmaf(wk, xv, ak[i]);
            av[i] = fmaf(wv, xv, av[i]);
        }
    }
    #pragma unroll
    for (int i = 0; i < 16; ++i) {
        int lloc = jb + i;
        qls[lloc * 64 + o] = f2bf(aq[i]);
        kls[lloc * 64 + o] = f2bf(ak[i]);
        vls[lloc * 64 + o] = f2bf(av[i]);
    }
    __syncthreads();

    // ---- fragment emission: 8 waves, each emits one Q, one K, one V frag ----
    const int w    = t >> 5;
    const int lane = t & 31;
    const int m    = lane & 15;
    const int half = lane >> 4;

    { // Q (A-frag) and K (B-frag): tile = w>>1 (local 16-tile), cc = w&1
        const int tl = w >> 1, cc = w & 1;
        u32 qu[8], ku[8];
        #pragma unroll
        for (int v = 0; v < 8; ++v) {
            int kA = (v < 4) ? (2 * v + 8 * half) : (16 + 2 * (v - 4) + 8 * half);
            qu[v] = *(const u32*)&qls[(tl * 16 + m) * 64 + (cc * 32 + kA)];
            int kB = (v < 4) ? (2 * v + 8 * half) : (16 + 2 * (v - 4) + 8 * half);
            ku[v] = *(const u32*)&kls[(tl * 16 + m) * 64 + (cc * 32 + kB)];
        }
        const int tglob = chunk * 4 + tl;
        u32* qd = qf + ((((bt * 64 + tglob) * 2 + cc) * 32 + lane) << 3);
        u32* kd = kf + ((((bt * 64 + tglob) * 2 + cc) * 32 + lane) << 3);
        ((uint4*)qd)[0] = make_uint4(qu[0], qu[1], qu[2], qu[3]);
        ((uint4*)qd)[1] = make_uint4(qu[4], qu[5], qu[6], qu[7]);
        ((uint4*)kd)[0] = make_uint4(ku[0], ku[1], ku[2], ku[3]);
        ((uint4*)kd)[1] = make_uint4(ku[4], ku[5], ku[6], ku[7]);
    }
    { // V (B-frag over 32-wide j tile): jl = w>>2, h = w&3
        const int jl = w >> 2, h = w & 3;
        const int cV = h * 16 + m;
        u32 vu[8];
        #pragma unroll
        for (int v = 0; v < 8; ++v) {
            int j0 = jl * 32 + ((v < 4) ? (2 * v + 8 * half) : (16 + 2 * (v - 4) + 8 * half));
            u32 lo = (u32)vls[j0 * 64 + cV];
            u32 hi = (u32)vls[(j0 + 1) * 64 + cV];
            vu[v] = lo | (hi << 16);
        }
        const int jglob = chunk * 2 + jl;
        u32* vd = vf + ((((bt * 32 + jglob) * 4 + h) * 32 + lane) << 3);
        ((uint4*)vd)[0] = make_uint4(vu[0], vu[1], vu[2], vu[3]);
        ((uint4*)vd)[1] = make_uint4(vu[4], vu[5], vu[6], vu[7]);
    }
}

// ---------------------------------------------------------------------------
// Kernel 2: flash attention. One 16-row query tile per wave, 8 waves/WG.
// grid.x = NBT * 8 ;  bt = blockIdx.x>>3, it = (blockIdx.x&7)*8 + wave
// Per 32-wide j tile: 4 WMMA (S=QK^T), permlane16 row-max (VALU),
// v_exp_f32, LDS relayout C->A frag, 1 WMMA (row sums = P*ones),
// 4 WMMA (O += P V^T).
// ---------------------------------------------------------------------------
__global__ __launch_bounds__(256)
void flash_attn_kernel(const u32* __restrict__ qf, const u32* __restrict__ kf,
                       const u32* __restrict__ vf, float* __restrict__ out)
{
    const int bt   = blockIdx.x >> 3;
    const int w    = threadIdx.x >> 5;
    const int lane = threadIdx.x & 31;
    const int it   = ((blockIdx.x & 7) << 3) + w;
    const int n    = lane & 15;
    const int half = lane >> 4;

    __shared__ u16 pl[8][16][32];   // per-wave P relayout scratch (8 KB)

    // load Q A-fragments (two 32-wide c chunks)
    FragU qa0, qa1;
    {
        const u32* qb = qf + ((((bt * 64 + it) * 2) * 32 + lane) << 3);
        *(uint4*)&qa0.u[0] = ((const uint4*)qb)[0];
        *(uint4*)&qa0.u[4] = ((const uint4*)qb)[1];
        const u32* qb1 = qb + (32 << 3);       // cc=1
        *(uint4*)&qa1.u[0] = ((const uint4*)qb1)[0];
        *(uint4*)&qa1.u[4] = ((const uint4*)qb1)[1];
    }

    // B-fragment of all-ones (bf16 1.0 pairs): row-sum reducer for P
    FragU ones;
    #pragma unroll
    for (int v = 0; v < 8; ++v) ones.u[v] = 0x3F803F80u;

    v8f acc[4];
    #pragma unroll
    for (int h = 0; h < 4; ++h) acc[h] = 0.0f;
    float mrun[8], lrun[8];
    #pragma unroll
    for (int r = 0; r < 8; ++r) { mrun[r] = -__builtin_inff(); lrun[r] = 0.0f; }

    for (int j32 = 0; j32 < 32; ++j32) {
        // ---- S = Q K^T over this 32-wide j tile (two 16x16 outputs) ----
        FragU kb[2][2];
        #pragma unroll
        for (int hh = 0; hh < 2; ++hh)
            #pragma unroll
            for (int cc = 0; cc < 2; ++cc) {
                const u32* kbp = kf + ((((bt * 64 + (2 * j32 + hh)) * 2 + cc) * 32 + lane) << 3);
                *(uint4*)&kb[hh][cc].u[0] = ((const uint4*)kbp)[0];
                *(uint4*)&kb[hh][cc].u[4] = ((const uint4*)kbp)[1];
            }
        if (j32 + 1 < 32) {
            __builtin_prefetch(kf + ((((bt * 64 + (2 * j32 + 2)) * 2) * 32 + lane) << 3), 0, 1);
            __builtin_prefetch(vf + ((((bt * 32 + (j32 + 1)) * 4) * 32 + lane) << 3), 0, 1);
        }

        v8f s0 = 0.0f, s1 = 0.0f;
        s0 = __builtin_amdgcn_wmma_f32_16x16x32_bf16(false, qa0.v, false, kb[0][0].v,
                                                     (short)0, s0, false, false);
        s0 = __builtin_amdgcn_wmma_f32_16x16x32_bf16(false, qa1.v, false, kb[0][1].v,
                                                     (short)0, s0, false, false);
        s1 = __builtin_amdgcn_wmma_f32_16x16x32_bf16(false, qa0.v, false, kb[1][0].v,
                                                     (short)0, s1, false, false);
        s1 = __builtin_amdgcn_wmma_f32_16x16x32_bf16(false, qa1.v, false, kb[1][1].v,
                                                     (short)0, s1, false, false);

        // ---- online softmax: row max (VALU permlane butterfly), exp, rescale ----
        float p0[8], p1[8];
        #pragma unroll
        for (int r = 0; r < 8; ++r) {
            float rm = rmax16(fmaxf(s0[r], s1[r]));
            float mn = fmaxf(mrun[r], rm);
            float sc = __expf(mrun[r] - mn);
            p0[r] = __expf(s0[r] - mn);
            p1[r] = __expf(s1[r] - mn);
            lrun[r] *= sc;
            mrun[r] = mn;
            acc[0][r] *= sc; acc[1][r] *= sc; acc[2][r] *= sc; acc[3][r] *= sc;
        }

        // ---- C-fragment P -> A-fragment P via per-wave LDS tile ----
        #pragma unroll
        for (int r = 0; r < 8; ++r) {
            int mrow = r + (half << 3);
            pl[w][mrow][n]      = f2bf(p0[r]);
            pl[w][mrow][n + 16] = f2bf(p1[r]);
        }
        asm volatile("s_wait_dscnt 0x0" ::: "memory");
        FragU pa;
        #pragma unroll
        for (int v = 0; v < 8; ++v) {
            int k = (v < 4) ? (2 * v + 8 * half) : (16 + 2 * (v - 4) + 8 * half);
            pa.u[v] = *(const u32*)&pl[w][n][k];
        }
        asm volatile("s_wait_dscnt 0x0" ::: "memory");

        // ---- row sums of P via WMMA against all-ones B (replaces shuffles) ----
        v8f rs = 0.0f;
        rs = __builtin_amdgcn_wmma_f32_16x16x32_bf16(false, pa.v, false, ones.v,
                                                     (short)0, rs, false, false);
        #pragma unroll
        for (int r = 0; r < 8; ++r) lrun[r] += rs[r];

        // ---- O += P V^T ----
        FragU vb[4];
        #pragma unroll
        for (int h = 0; h < 4; ++h) {
            const u32* vbp = vf + ((((bt * 32 + j32) * 4 + h) * 32 + lane) << 3);
            *(uint4*)&vb[h].u[0] = ((const uint4*)vbp)[0];
            *(uint4*)&vb[h].u[4] = ((const uint4*)vbp)[1];
        }
        #pragma unroll
        for (int h = 0; h < 4; ++h)
            acc[h] = __builtin_amdgcn_wmma_f32_16x16x32_bf16(false, pa.v, false, vb[h].v,
                                                             (short)0, acc[h], false, false);
    }

    // ---- epilogue: normalize rows, store out[bt][c][i] ----
    float inv[8];
    #pragma unroll
    for (int r = 0; r < 8; ++r) inv[r] = 1.0f / lrun[r];
    #pragma unroll
    for (int h = 0; h < 4; ++h) {
        const int c = h * 16 + n;
        float* po = out + ((bt * CC + c) << 10) + it * 16 + (half << 3);
        float4 a = make_float4(acc[h][0] * inv[0], acc[h][1] * inv[1],
                               acc[h][2] * inv[2], acc[h][3] * inv[3]);
        float4 b = make_float4(acc[h][4] * inv[4], acc[h][5] * inv[5],
                               acc[h][6] * inv[6], acc[h][7] * inv[7]);
        *(float4*)(po)     = a;
        *(float4*)(po + 4) = b;
    }
}

// ---------------------------------------------------------------------------
extern "C" void kernel_launch(void* const* d_in, const int* in_sizes, int n_in,
                              void* d_out, int out_size, void* d_ws, size_t ws_size,
                              hipStream_t stream)
{
    const float* x  = (const float*)d_in[0];
    const float* Wq = (const float*)d_in[1];
    const float* bq = (const float*)d_in[2];
    const float* Wk = (const float*)d_in[3];
    const float* bk = (const float*)d_in[4];
    const float* Wv = (const float*)d_in[5];
    const float* bv = (const float*)d_in[6];

    // workspace: Q frags (8MB) | K frags (8MB) | V frags (8MB)
    u32* qf = (u32*)d_ws;
    u32* kf = qf + (1u << 21);
    u32* vf = kf + (1u << 21);

    // kernel1 dynamic LDS: 3*16KB weights + 768B bias + 16KB x + 3*8KB bf16 qkv
    const size_t smem1 = (4096 * 3 + 192 + 4096) * sizeof(float) + 4096 * 3 * sizeof(u16);

    qkv_pack_kernel<<<NBT * 16, 256, smem1, stream>>>(x, Wq, bq, Wk, bk, Wv, bv, qf, kf, vf);
    flash_attn_kernel<<<NBT * 8, 256, 0, stream>>>(qf, kf, vf, (float*)d_out);
}
